// HyperMultiHeadAttention_80848464380550
// MI455X (gfx1250) — matrix operand order repaired and checked by reading, not compiled, not run on hardware
//
#include <hip/hip_runtime.h>
#include <hip/hip_bf16.h>
#include <stdint.h>
#include <stddef.h>

typedef __bf16 bf16_t;
typedef __attribute__((ext_vector_type(16))) __bf16 v16bf;
typedef __attribute__((ext_vector_type(8)))  float  v8f;

namespace {

constexpr int B = 4, S = 2048, E = 1024, H = 16, D = 64;
constexpr int BS = B * S;      // 8192 rows
constexpr int E3 = 3 * E;      // 3072
constexpr float NEG = -1e9f;

union FragU { v16bf v; uint4 u[2]; };

// A-matrix fragment (16x32 bf16): lane%16 -> M row, e<8 -> K=k0+half*8+e,
// e>=8 -> K=k0+16+half*8+(e-8). Two 16B loads per lane.
__device__ __forceinline__ v16bf load_a_frag(const bf16_t* base, int ld,
                                             int row, int half, int k0) {
  const bf16_t* p = base + (size_t)row * ld + k0 + half * 8;
  FragU f;
  f.u[0] = *(const uint4*)(p);
  f.u[1] = *(const uint4*)(p + 16);
  return f.v;
}

// B-matrix fragment (32x16 bf16, source stored N-major with K contiguous):
// lane%16 -> N col, e<8 -> K=k0+half*8+e, e>=8 -> K=k0+16+half*8+(e-8).
__device__ __forceinline__ v16bf load_b_frag(const bf16_t* base, int ld,
                                             int col, int half, int k0) {
  const bf16_t* p = base + (size_t)col * ld + k0 + half * 8;
  FragU f;
  f.u[0] = *(const uint4*)(p);
  f.u[1] = *(const uint4*)(p + 16);
  return f.v;
}

__device__ __forceinline__ v8f wmma_bf16(v16bf a, v16bf b, v8f c) {
  return __builtin_amdgcn_wmma_f32_16x16x32_bf16(false, a, false, b,
                                                 (short)0, c, false, false);
}

// Async global->LDS copy of 16 bytes per lane (CDNA5, tracked by ASYNCcnt).
__device__ __forceinline__ void async_copy_b128(const void* gaddr,
                                                void* lds_ptr) {
  uint32_t lds = (uint32_t)(uintptr_t)lds_ptr;  // low 32 bits = LDS offset
  asm volatile("global_load_async_to_lds_b128 %0, %1, off"
               :: "v"(lds), "v"(gaddr) : "memory");
}
__device__ __forceinline__ void async_wait0() {
  asm volatile("s_wait_asynccnt 0" ::: "memory");
}

__global__ void cvt_f32_bf16(const float* __restrict__ in,
                             bf16_t* __restrict__ out, int n) {
  int i = blockIdx.x * blockDim.x + threadIdx.x;
  if (i < n) out[i] = (bf16_t)in[i];
}

// ---------------------------------------------------------------------------
// QKV projection: C[BS,3E] = Xb[BS,E] * Win^T + bias. One wave -> 32x64 tile
// (2 A-fragments, 8 accumulators; each B-fragment feeds 2 WMMAs).
// Q scaled by 1/sqrt(D), written [B,H,S,D]; K written [B,H,S,D];
// V written TRANSPOSED [B,H,D,S] for the attention P*V B-fragments.
// ---------------------------------------------------------------------------
__global__ __launch_bounds__(256) void qkv_gemm_kernel(
    const bf16_t* __restrict__ xb, const bf16_t* __restrict__ wb,
    const float* __restrict__ bias,
    bf16_t* __restrict__ qo, bf16_t* __restrict__ ko,
    bf16_t* __restrict__ vTo) {
  const int gw   = blockIdx.x * 8 + (threadIdx.x >> 5);
  const int lane = threadIdx.x & 31;
  const int ln = lane & 15, half = lane >> 4;
  const int ntile = gw % (E3 / 64);   // 48
  const int mtile = gw / (E3 / 64);   // 256
  const int m0 = mtile * 32, n0 = ntile * 64;
  const bf16_t* arow = xb + (size_t)m0 * E;

  v8f acc[2][4] = {};
  for (int k0 = 0; k0 < E; k0 += 32) {
    __builtin_prefetch(arow + (size_t)ln * E + k0 + 64, 0, 1);
    v16bf a0 = load_a_frag(arow, E, ln, half, k0);
    v16bf a1 = load_a_frag(arow + (size_t)16 * E, E, ln, half, k0);
#pragma unroll
    for (int t = 0; t < 4; ++t) {
      v16bf b = load_b_frag(wb + (size_t)(n0 + t * 16) * E, E, ln, half, k0);
      acc[0][t] = wmma_bf16(a0, b, acc[0][t]);
      acc[1][t] = wmma_bf16(a1, b, acc[1][t]);
    }
  }

  const int bb = m0 / S;
  const int s0 = m0 % S;
#pragma unroll
  for (int t = 0; t < 4; ++t) {
    const int n = n0 + t * 16 + ln;
    const float bv = bias[n];
    const int sect = n >> 10;          // 0:q 1:k 2:v (uniform per wave)
    const int h = (n & 1023) >> 6;
    const int d = n & 63;
#pragma unroll
    for (int i = 0; i < 2; ++i)
#pragma unroll
      for (int r = 0; r < 8; ++r) {
        const int s = s0 + i * 16 + r + half * 8;
        const float val = acc[i][t][r] + bv;
        if (sect == 0)
          qo[(((size_t)bb * H + h) * S + s) * D + d] = (bf16_t)(val * 0.125f);
        else if (sect == 1)
          ko[(((size_t)bb * H + h) * S + s) * D + d] = (bf16_t)val;
        else
          vTo[(((size_t)bb * H + h) * D + d) * S + s] = (bf16_t)val;
      }
  }
}

// ---------------------------------------------------------------------------
// Flash attention. Block = 8 waves = 8 consecutive q-tiles of ONE (b,h) head.
// Per Sk block of 32 keys the block cooperatively async-stages the K tile
// (32x64) and V^T tile (64x32) into LDS (global_load_async_to_lds_b128,
// ASYNCcnt), then every wave runs QK^T / online softmax / P*V from LDS.
// P goes through a per-wave LDS scratch to convert C-layout -> A-fragments.
// ---------------------------------------------------------------------------
__global__ __launch_bounds__(256) void attn_kernel(
    const bf16_t* __restrict__ q, const bf16_t* __restrict__ k,
    const bf16_t* __restrict__ vT, const int* __restrict__ mask,
    bf16_t* __restrict__ ctx) {
  const int w    = threadIdx.x >> 5;
  const int lane = threadIdx.x & 31;
  const int ln = lane & 15, half = lane >> 4;
  const int bh  = blockIdx.x >> 4;            // 0..63 (b,h)
  const int qt  = (blockIdx.x & 15) * 8 + w;  // 0..127 query tile
  const int bb = bh >> 4, h = bh & 15;

  const bf16_t* qb = q + ((size_t)bh * S + qt * 16) * D;
  const bf16_t* kb = k + (size_t)bh * S * D;
  const bf16_t* vb = vT + (size_t)bh * D * S;
  const int* mb = mask + bb * S;

  __shared__ bf16_t ksh[32][64];             // 4 KB K tile   [key][d]
  __shared__ bf16_t vsh[64][32];             // 4 KB V^T tile [d][key]
  __shared__ bf16_t pbuf[8][16][32];         // 8 KB P scratch, one per wave
  bf16_t(*pt)[32] = pbuf[w];

  const v16bf qa0 = load_a_frag(qb, D, ln, half, 0);
  const v16bf qa1 = load_a_frag(qb, D, ln, half, 32);

  v8f acc[4] = {};
  float rmax[8], rsum[8];
#pragma unroll
  for (int r = 0; r < 8; ++r) { rmax[r] = -3.0e38f; rsum[r] = 0.0f; }

  const int tid  = threadIdx.x;
  const int krow = tid >> 3, kseg = tid & 7;  // 32 rows x 8 x 16B = 4 KB
  const int vrow = tid >> 2, vseg = tid & 3;  // 64 rows x 4 x 16B = 4 KB

  for (int sk0 = 0; sk0 < S; sk0 += 32) {
    // ---- cooperative async stage of K / V^T tiles into LDS ----
    async_copy_b128(kb + (size_t)(sk0 + krow) * D + kseg * 8,
                    &ksh[krow][kseg * 8]);
    async_copy_b128(vb + (size_t)vrow * S + sk0 + vseg * 8,
                    &vsh[vrow][vseg * 8]);
    async_wait0();
    __syncthreads();

    // ---- scores: 16x32 = Q(16x64) * K^T, two 16x16 WMMA tiles ----
    v8f sc[2];
#pragma unroll
    for (int t = 0; t < 2; ++t) {
      v16bf b0 = load_b_frag(&ksh[0][0], 64, t * 16 + ln, half, 0);
      v16bf b1 = load_b_frag(&ksh[0][0], 64, t * 16 + ln, half, 32);
      v8f c = {};
      c = wmma_bf16(qa0, b0, c);
      c = wmma_bf16(qa1, b1, c);
      const bool pad = (mb[sk0 + t * 16 + ln] == 0);  // key padding mask
#pragma unroll
      for (int r = 0; r < 8; ++r) c[r] = pad ? NEG : c[r];
      sc[t] = c;
    }
    // ---- online softmax: row m = r + half*8; N over the 16-lane group ----
#pragma unroll
    for (int r = 0; r < 8; ++r) {
      float v = fmaxf(sc[0][r], sc[1][r]);
      v = fmaxf(v, __shfl_xor(v, 1));
      v = fmaxf(v, __shfl_xor(v, 2));
      v = fmaxf(v, __shfl_xor(v, 4));
      v = fmaxf(v, __shfl_xor(v, 8));
      const float mnew  = fmaxf(rmax[r], v);
      const float scale = __expf(rmax[r] - mnew);
      rmax[r] = mnew;
      const float e0 = __expf(sc[0][r] - mnew);
      const float e1 = __expf(sc[1][r] - mnew);
      sc[0][r] = e0;
      sc[1][r] = e1;
      float rs = e0 + e1;
      rs += __shfl_xor(rs, 1);
      rs += __shfl_xor(rs, 2);
      rs += __shfl_xor(rs, 4);
      rs += __shfl_xor(rs, 8);
      rsum[r] = rsum[r] * scale + rs;
#pragma unroll
      for (int t = 0; t < 4; ++t) acc[t][r] *= scale;
    }
    // ---- P (C layout) -> per-wave LDS -> A-fragment ----
#pragma unroll
    for (int t = 0; t < 2; ++t)
#pragma unroll
      for (int r = 0; r < 8; ++r)
        pt[r + half * 8][t * 16 + ln] = (bf16_t)sc[t][r];
    v16bf pa = load_a_frag(&pt[0][0], 32, ln, half, 0);
    // ---- acc += P(16x32) * V(32x64), B-fragments from LDS V^T tile ----
#pragma unroll
    for (int t = 0; t < 4; ++t) {
      v16bf vf = load_b_frag(&vsh[0][0], 32, t * 16 + ln, half, 0);
      acc[t] = wmma_bf16(pa, vf, acc[t]);
    }
    __syncthreads();  // protect ksh/vsh before next async stage
  }
  // ---- normalize, store ctx [B, S, H*D] bf16 ----
#pragma unroll
  for (int r = 0; r < 8; ++r) {
    const float inv = 1.0f / rsum[r];
    const int sq = qt * 16 + r + half * 8;
#pragma unroll
    for (int t = 0; t < 4; ++t)
      ctx[((size_t)bb * S + sq) * E + h * D + t * 16 + ln] =
          (bf16_t)(acc[t][r] * inv);
  }
}

// ---------------------------------------------------------------------------
// Output projection + bias + residual: out = x + ctx * Wout^T + bout  (f32)
// One wave -> 32x64 tile.
// ---------------------------------------------------------------------------
__global__ __launch_bounds__(256) void out_proj_kernel(
    const bf16_t* __restrict__ ctx, const bf16_t* __restrict__ wo,
    const float* __restrict__ bias, const float* __restrict__ x,
    float* __restrict__ out) {
  const int gw   = blockIdx.x * 8 + (threadIdx.x >> 5);
  const int lane = threadIdx.x & 31;
  const int ln = lane & 15, half = lane >> 4;
  const int ntile = gw % (E / 64);    // 16
  const int mtile = gw / (E / 64);    // 256
  const int m0 = mtile * 32, n0 = ntile * 64;
  const bf16_t* arow = ctx + (size_t)m0 * E;

  v8f acc[2][4] = {};
  for (int k0 = 0; k0 < E; k0 += 32) {
    __builtin_prefetch(arow + (size_t)ln * E + k0 + 64, 0, 1);
    v16bf a0 = load_a_frag(arow, E, ln, half, k0);
    v16bf a1 = load_a_frag(arow + (size_t)16 * E, E, ln, half, k0);
#pragma unroll
    for (int t = 0; t < 4; ++t) {
      v16bf b = load_b_frag(wo + (size_t)(n0 + t * 16) * E, E, ln, half, k0);
      acc[0][t] = wmma_bf16(a0, b, acc[0][t]);
      acc[1][t] = wmma_bf16(a1, b, acc[1][t]);
    }
  }
#pragma unroll
  for (int t = 0; t < 4; ++t) {
    const int n = n0 + t * 16 + ln;
    const float bv = bias[n];
#pragma unroll
    for (int i = 0; i < 2; ++i)
#pragma unroll
      for (int r = 0; r < 8; ++r) {
        const size_t idx = (size_t)(m0 + i * 16 + r + half * 8) * E + n;
        out[idx] = acc[i][t][r] + bv + x[idx];
      }
  }
}

}  // namespace

extern "C" void kernel_launch(void* const* d_in, const int* in_sizes, int n_in,
                              void* d_out, int out_size, void* d_ws,
                              size_t ws_size, hipStream_t stream) {
  (void)in_sizes; (void)n_in; (void)out_size; (void)ws_size;
  const float* x     = (const float*)d_in[0];  // [B,S,E]
  const float* w_in  = (const float*)d_in[1];  // [3E,E]
  const float* b_in  = (const float*)d_in[2];  // [3E]
  const float* w_out = (const float*)d_in[3];  // [E,E]
  const float* b_out = (const float*)d_in[4];  // [E]
  const int*   mask  = (const int*)d_in[5];    // [B,S]
  float* out = (float*)d_out;

  // Workspace carve-up (~92 MB total, 256B aligned slices).
  char* ws = (char*)d_ws;
  size_t off = 0;
  auto carve = [&](size_t bytes) {
    void* p = ws + off;
    off += (bytes + 255) & ~(size_t)255;
    return p;
  };
  bf16_t* xb  = (bf16_t*)carve((size_t)BS * E * 2);   // x in bf16
  bf16_t* wib = (bf16_t*)carve((size_t)E3 * E * 2);   // Win bf16
  bf16_t* wob = (bf16_t*)carve((size_t)E * E * 2);    // Wout bf16
  bf16_t* qb  = (bf16_t*)carve((size_t)BS * E * 2);   // Q [B,H,S,D]
  bf16_t* kb  = (bf16_t*)carve((size_t)BS * E * 2);   // K [B,H,S,D]
  bf16_t* vTb = (bf16_t*)carve((size_t)BS * E * 2);   // V^T [B,H,D,S]
  bf16_t* cb  = (bf16_t*)carve((size_t)BS * E * 2);   // ctx [B,S,E]

  cvt_f32_bf16<<<(BS * E + 255) / 256, 256, 0, stream>>>(x, xb, BS * E);
  cvt_f32_bf16<<<(E3 * E + 255) / 256, 256, 0, stream>>>(w_in, wib, E3 * E);
  cvt_f32_bf16<<<(E * E + 255) / 256, 256, 0, stream>>>(w_out, wob, E * E);

  // 256 M-tiles x 48 N-tiles = 12288 waves / 8 per block
  qkv_gemm_kernel<<<(BS / 32) * (E3 / 64) / 8, 256, 0, stream>>>(
      xb, wib, b_in, qb, kb, vTb);

  // 64 (b,h) x 16 blocks of 8 q-tiles = 1024 blocks
  attn_kernel<<<(B * H) * (S / 16) / 8, 256, 0, stream>>>(qb, kb, vTb, mask, cb);

  // 256 M-tiles x 16 N-tiles = 4096 waves / 8 per block
  out_proj_kernel<<<(BS / 32) * (E / 64) / 8, 256, 0, stream>>>(
      cb, wob, b_out, x, out);
}
